// LinkPrediction_22376779612762
// MI455X (gfx1250) — matrix operand order repaired and checked by reading, not compiled, not run on hardware
//
#include <hip/hip_runtime.h>

// ---------------------------------------------------------------------------
// MI455X link-prediction MLP — v2.
// Roofline: ~157 GFLOP of GEMM; x/w1/w2 are L2-resident (25.6MB + 320KB).
// HBM traffic ~tens of MB -> pure matrix-core problem, routed through
// v_wmma_f32_16x16x32_f16.
//
// v2 changes (B-operand bandwidth was the bottleneck in v1):
//  * 32 edges per wave (two 16-row A tiles share every B-fragment load):
//    B traffic halved to ~6KB/edge; 1 b128-pair feeds 2 WMMAs.
//  * LayerNorm stats computed from accumulator registers (per-lane static row
//    ownership: row = mt*16 + v + 8*(lane>>4)), merged with shfl_xor across
//    the 16-lane half. h1 hits LDS exactly once (f16); layer-2 LN and the
//    final 128-dot never touch LDS.
//  * Layer-1 A fragments loaded directly from xh (two b128/lane), no staged
//    gather, no f32 scratch: LDS/wave drops 24KB -> 16.4KB.
// ---------------------------------------------------------------------------

typedef __attribute__((ext_vector_type(16))) _Float16 v16h;
typedef __attribute__((ext_vector_type(8)))  float    v8f;

#define DD   128      // per-node feature dim
#define H1   256      // hidden 1
#define H2   128      // hidden 2
#define TM   32       // edges per wave tile (2 x 16-row WMMA tiles)
#define WPB  2        // waves per block -> ~33KB LDS

union Frag16 { v16h h; uint4 u[2]; };
union Oct16  { uint4 u; _Float16 hh[8]; };

// ---------------- prep kernels ----------------

__global__ void k_cvt_x(const float* __restrict__ x, _Float16* __restrict__ xh, int n) {
  int i = blockIdx.x * blockDim.x + threadIdx.x;
  if (i < n) xh[i] = (_Float16)x[i];
}

// Pack row-major f32 [K=256, ncols] into WMMA B fragments (f16).
// Fragment f = kb*nJ + jn covers K = kb*32..+32, N = jn*16..+16.
// Lane L owns column jn*16+(L&15); half i holds K = kb*32 + (L>>4)*8 +
// (i>>3)*16 + (i&7) (16-bit B layout, table order matching the A fragments).
__global__ void k_pack_w(const float* __restrict__ w, _Float16* __restrict__ wp,
                         int ncols, int nJ) {
  int f = blockIdx.x;
  int kb = f / nJ, jn = f % nJ;
  int lane = threadIdx.x;
  int col   = jn * 16 + (lane & 15);
  int kbase = kb * 32 + (lane >> 4) * 8;
#pragma unroll
  for (int i = 0; i < 16; ++i)
    wp[(size_t)f * 512 + lane * 16 + i] = (_Float16)w[(size_t)(kbase + (i >> 3) * 16 + (i & 7)) * ncols + col];
}

// ---------------- main edge kernel ----------------

__global__ __launch_bounds__(WPB * 32)
void k_edge_mlp(const _Float16* __restrict__ xh,
                const _Float16* __restrict__ w1p,
                const _Float16* __restrict__ w2p,
                const int* __restrict__ ei,  const int* __restrict__ nei,
                const float* __restrict__ b1, const float* __restrict__ g1,
                const float* __restrict__ be1,
                const float* __restrict__ b2, const float* __restrict__ g2,
                const float* __restrict__ be2,
                const float* __restrict__ w3, const float* __restrict__ b3,
                float* __restrict__ out,
                int Epos, int total, int numTiles)
{
  __shared__ __align__(16) _Float16 h1S_[WPB][TM * H1];   // 32x256 f16 per wave
  __shared__ float muS_[WPB][TM], rsS_[WPB][TM], zS_[WPB][TM];

  const int wave = threadIdx.x >> 5;
  const int lane = threadIdx.x & 31;
  const int tile = blockIdx.x * WPB + wave;
  if (tile >= numTiles) return;        // wave-private LDS slices: no barriers

  _Float16* h1S = h1S_[wave];
  float* muS = muS_[wave];
  float* rsS = rsS_[wave];
  float* zS  = zS_[wave];

  const int kg  = lane >> 4;           // K-group / row-half selector
  const int l15 = lane & 15;

  // ---- fetch 32 edge endpoints (one per lane) ----
  int srcN, dstN;
  {
    int e = tile * TM + lane; if (e >= total) e = total - 1;
    if (e < Epos) { srcN = ei[e];              dstN = ei[Epos + e]; }
    else { int e2 = e - Epos; int En = total - Epos;
           srcN = nei[e2];                     dstN = nei[En + e2]; }
  }
  // node row base pointers for the two A tiles this lane helps load
  size_t sNode[2], dNode[2];
#pragma unroll
  for (int mt = 0; mt < 2; ++mt) {
    sNode[mt] = (size_t)__shfl(srcN, mt * 16 + l15, 32) * DD;
    dNode[mt] = (size_t)__shfl(dstN, mt * 16 + l15, 32) * DD;
  }

  const v8f vzero = {};
  float s1[2][8], q1[2][8];
#pragma unroll
  for (int mt = 0; mt < 2; ++mt)
#pragma unroll
    for (int v = 0; v < 8; ++v) { s1[mt][v] = 0.f; q1[mt][v] = 0.f; }

  // ================= layer 1: [32,256] @ w1[256,256] =================
  // N processed in two halves of 128 to bound accumulator registers.
  for (int nh = 0; nh < 2; ++nh) {
    v8f acc[2][8];
#pragma unroll
    for (int mt = 0; mt < 2; ++mt)
#pragma unroll
      for (int j = 0; j < 8; ++j) acc[mt][j] = vzero;

    for (int kb = 0; kb < H1 / 32; ++kb) {
      // A fragments straight from xh: feat col = kb*32+.. ; kb<4 -> src half
      Frag16 a[2];
      int off = (kb & 3) * 32 + kg * 8;
#pragma unroll
      for (int mt = 0; mt < 2; ++mt) {
        const uint4* ap = (const uint4*)(xh + (kb < 4 ? sNode[mt] : dNode[mt]) + off);
        a[mt].u[0] = ap[0];       // K = kb*32 + kg*8 + [0..8)
        a[mt].u[1] = ap[2];       // K = kb*32 + 16 + kg*8 + [0..8)
      }
#pragma unroll
      for (int jn = 0; jn < 8; ++jn) {
        Frag16 b;
        const uint4* bp =
            (const uint4*)(w1p + ((size_t)(kb * 16 + nh * 8 + jn) * 32 + lane) * 16);
        b.u[0] = bp[0]; b.u[1] = bp[1];
        acc[0][jn] = __builtin_amdgcn_wmma_f32_16x16x32_f16(
            false, a[0].h, false, b.h, (short)0, acc[0][jn], false, false);
        acc[1][jn] = __builtin_amdgcn_wmma_f32_16x16x32_f16(
            false, a[1].h, false, b.h, (short)0, acc[1][jn], false, false);
      }
    }
    // epilogue: +b1, accumulate LN stats in regs, spill pre-LN h1 as f16.
    // C layout: VGPR v -> row mt*16 + v + 8*kg, col n = nh*128 + jn*16 + l15.
#pragma unroll
    for (int jn = 0; jn < 8; ++jn) {
      int n = nh * 128 + jn * 16 + l15;
      float bv = b1[n];
#pragma unroll
      for (int mt = 0; mt < 2; ++mt)
#pragma unroll
        for (int v = 0; v < 8; ++v) {
          float t = acc[mt][jn][v] + bv;
          s1[mt][v] += t; q1[mt][v] += t * t;
          h1S[(mt * 16 + v + 8 * kg) * H1 + n] = (_Float16)t;
        }
    }
  }

  // ---- LN1 stats: reduce across the 16-lane half, publish per-row ----
#pragma unroll
  for (int mt = 0; mt < 2; ++mt)
#pragma unroll
    for (int v = 0; v < 8; ++v) {
#pragma unroll
      for (int d = 1; d < 16; d <<= 1) {
        s1[mt][v] += __shfl_xor(s1[mt][v], d, 32);
        q1[mt][v] += __shfl_xor(q1[mt][v], d, 32);
      }
    }
  if (l15 == 0) {
#pragma unroll
    for (int mt = 0; mt < 2; ++mt)
#pragma unroll
      for (int v = 0; v < 8; ++v) {
        int row = mt * 16 + v + 8 * kg;
        float mu = s1[mt][v] * (1.f / H1);
        float var = q1[mt][v] * (1.f / H1) - mu * mu;
        muS[row] = mu;
        rsS[row] = rsqrtf(var + 1e-5f);
      }
  }

  // ---- normalize + affine + relu in place (8 halves per iteration) ----
  for (int o = lane; o < TM * H1 / 8; o += 32) {
    int base = o * 8;
    int m = base >> 8, nb = base & (H1 - 1);
    float mu = muS[m], rs = rsS[m];
    Oct16 p; p.u = *(const uint4*)(h1S + base);
#pragma unroll
    for (int i = 0; i < 8; ++i) {
      int n = nb + i;
      float v = ((float)p.hh[i] - mu) * rs * g1[n] + be1[n];
      p.hh[i] = (_Float16)(v > 0.f ? v : 0.f);
    }
    *(uint4*)(h1S + base) = p.u;
  }

  // ================= layer 2: [32,256] @ w2[256,128] =================
  v8f acc2[2][8];
#pragma unroll
  for (int mt = 0; mt < 2; ++mt)
#pragma unroll
    for (int j = 0; j < 8; ++j) acc2[mt][j] = vzero;

  for (int kb = 0; kb < H1 / 32; ++kb) {
    Frag16 a[2];
#pragma unroll
    for (int mt = 0; mt < 2; ++mt) {
      const uint4* ap = (const uint4*)(h1S + (mt * 16 + l15) * H1 + kb * 32 + kg * 8);
      a[mt].u[0] = ap[0];
      a[mt].u[1] = ap[2];
    }
#pragma unroll
    for (int jn = 0; jn < 8; ++jn) {
      Frag16 b;
      const uint4* bp = (const uint4*)(w2p + ((size_t)(kb * 8 + jn) * 32 + lane) * 16);
      b.u[0] = bp[0]; b.u[1] = bp[1];
      acc2[0][jn] = __builtin_amdgcn_wmma_f32_16x16x32_f16(
          false, a[0].h, false, b.h, (short)0, acc2[0][jn], false, false);
      acc2[1][jn] = __builtin_amdgcn_wmma_f32_16x16x32_f16(
          false, a[1].h, false, b.h, (short)0, acc2[1][jn], false, false);
    }
  }

  // ---- LN2 entirely in registers (reuse s1/q1 as s2/q2) ----
  float b2v[8];
#pragma unroll
  for (int jn = 0; jn < 8; ++jn) b2v[jn] = b2[jn * 16 + l15];
#pragma unroll
  for (int mt = 0; mt < 2; ++mt)
#pragma unroll
    for (int v = 0; v < 8; ++v) { s1[mt][v] = 0.f; q1[mt][v] = 0.f; }
#pragma unroll
  for (int jn = 0; jn < 8; ++jn)
#pragma unroll
    for (int mt = 0; mt < 2; ++mt)
#pragma unroll
      for (int v = 0; v < 8; ++v) {
        float t = acc2[mt][jn][v] + b2v[jn];
        s1[mt][v] += t; q1[mt][v] += t * t;
      }
#pragma unroll
  for (int mt = 0; mt < 2; ++mt)
#pragma unroll
    for (int v = 0; v < 8; ++v) {
#pragma unroll
      for (int d = 1; d < 16; d <<= 1) {
        s1[mt][v] += __shfl_xor(s1[mt][v], d, 32);
        q1[mt][v] += __shfl_xor(q1[mt][v], d, 32);
      }
      float mu = s1[mt][v] * (1.f / H2);
      float var = q1[mt][v] * (1.f / H2) - mu * mu;
      s1[mt][v] = mu;                      // s1 now holds mu2
      q1[mt][v] = rsqrtf(var + 1e-5f);     // q1 now holds rs2
    }

  // ---- layer 3: normalized-relu dot with w3, reduce, sigmoid ----
  float g2v[8], be2v[8], w3v[8];
#pragma unroll
  for (int jn = 0; jn < 8; ++jn) {
    int n = jn * 16 + l15;
    g2v[jn] = g2[n]; be2v[jn] = be2[n]; w3v[jn] = w3[n];
  }
  float zp[2][8];
#pragma unroll
  for (int mt = 0; mt < 2; ++mt)
#pragma unroll
    for (int v = 0; v < 8; ++v) zp[mt][v] = 0.f;
#pragma unroll
  for (int jn = 0; jn < 8; ++jn)
#pragma unroll
    for (int mt = 0; mt < 2; ++mt)
#pragma unroll
      for (int v = 0; v < 8; ++v) {
        float t = acc2[mt][jn][v] + b2v[jn];
        float val = (t - s1[mt][v]) * q1[mt][v] * g2v[jn] + be2v[jn];
        val = val > 0.f ? val : 0.f;
        zp[mt][v] += val * w3v[jn];
      }
#pragma unroll
  for (int mt = 0; mt < 2; ++mt)
#pragma unroll
    for (int v = 0; v < 8; ++v) {
#pragma unroll
      for (int d = 1; d < 16; d <<= 1)
        zp[mt][v] += __shfl_xor(zp[mt][v], d, 32);
    }
  if (l15 == 0) {
#pragma unroll
    for (int mt = 0; mt < 2; ++mt)
#pragma unroll
      for (int v = 0; v < 8; ++v)
        zS[mt * 16 + v + 8 * kg] = zp[mt][v];
  }

  {
    int e = tile * TM + lane;
    if (e < total) {
      float z = zS[lane] + b3[0];
      out[e]         = 1.f / (1.f + __expf(-z));
      out[total + e] = (e < Epos) ? 1.f : 0.f;
    }
  }
}

// ---------------- host launcher ----------------

extern "C" void kernel_launch(void* const* d_in, const int* in_sizes, int n_in,
                              void* d_out, int out_size, void* d_ws, size_t ws_size,
                              hipStream_t stream) {
  const float* x   = (const float*)d_in[0];
  const int*   ei  = (const int*)  d_in[1];   // [2, Epos] flattened
  const int*   nei = (const int*)  d_in[3];   // [2, Eneg]
  const float* w1  = (const float*)d_in[4];
  const float* b1  = (const float*)d_in[5];
  const float* g1  = (const float*)d_in[6];
  const float* be1 = (const float*)d_in[7];
  const float* w2  = (const float*)d_in[8];
  const float* b2  = (const float*)d_in[9];
  const float* g2  = (const float*)d_in[10];
  const float* be2 = (const float*)d_in[11];
  const float* w3  = (const float*)d_in[12];
  const float* b3  = (const float*)d_in[13];
  float* out = (float*)d_out;

  const int N     = in_sizes[0] / DD;
  const int Epos  = in_sizes[1] / 2;
  const int Eneg  = in_sizes[3] / 2;
  const int total = Epos + Eneg;

  // workspace layout (f16): xh [N*128] | w1p [256*256] | w2p [256*128]
  _Float16* xh  = (_Float16*)d_ws;
  _Float16* w1p = xh + (size_t)N * DD;
  _Float16* w2p = w1p + (size_t)H1 * H1;

  int nx = N * DD;
  k_cvt_x<<<(nx + 255) / 256, 256, 0, stream>>>(x, xh, nx);
  k_pack_w<<<(H1 / 32) * (H1 / 16), 32, 0, stream>>>(w1, w1p, H1, H1 / 16); // 128 frags
  k_pack_w<<<(H1 / 32) * (H2 / 16), 32, 0, stream>>>(w2, w2p, H2, H2 / 16); //  64 frags

  int numTiles = (total + TM - 1) / TM;
  int blocks   = (numTiles + WPB - 1) / WPB;
  k_edge_mlp<<<blocks, WPB * 32, 0, stream>>>(
      xh, w1p, w2p, ei, nei, b1, g1, be1, b2, g2, be2, w3, b3,
      out, Epos, total, numTiles);
}
